// HierarchicalClusterLocalAttention_26929444946509
// MI455X (gfx1250) — matrix operand order repaired and checked by reading, not compiled, not run on hardware
//
#include <hip/hip_runtime.h>
#include <hip/hip_bf16.h>

// ---------------------------------------------------------------------------
// HierarchicalClusterLocalAttention for MI455X (gfx1250, wave32, WMMA)
// All large GEMMs use v_wmma_f32_16x16x32_bf16 (bf16 A/B, f32 accumulate).
// ---------------------------------------------------------------------------

typedef __bf16 bf16_t;
typedef __attribute__((ext_vector_type(16))) __bf16 v16bf;
typedef __attribute__((ext_vector_type(8)))  __bf16 v8bf;
typedef __attribute__((ext_vector_type(8)))  float  v8f;

constexpr int LT   = 4096;   // tokens
constexpr int CD   = 384;    // hidden
constexpr int C3   = 1152;   // 3*hidden
constexpr int NH   = 8;      // heads
constexpr int HD   = 48;     // head dim
constexpr int NCL  = 16;     // clusters (4096/256)
constexpr int KNNK = 10;     // top-k
constexpr float LNEPS = 1e-5f;
constexpr float SCALE = 0.14433756729740645f; // 1/sqrt(48)

union Frag { v16bf v; v8bf h[2]; };

// Load 8 contiguous bf16 (16B); zero beyond K (K always multiple of 8 here).
__device__ __forceinline__ v8bf load_chunk(const bf16_t* __restrict__ rowp, int k, int K) {
  if (k + 8 <= K) return *(const v8bf*)(rowp + k);
  v8bf z = {};
  return z;
}

// One 16x16 f32 tile of C = A[M,K] * Bt[N,K]^T using WMMA bf16.
// A-fragment layout (ISA 7.12.2, 16-bit A 16x32): lane = m + 16*half,
//   elems 0..7 -> K = k0+8*half+0..7 ; elems 8..15 -> K = k0+16+8*half+0..7
// B-fragment: lane = n + 16*half, same K order as A (elems 0..7 -> k0+8*half, 8..15 -> k0+16+8*half) of col n
__device__ __forceinline__ v8f wmma_tile_nt(const bf16_t* __restrict__ A, int lda,
                                            const bf16_t* __restrict__ Bt, int ldb,
                                            int row0, int col0, int K) {
  const int lane = threadIdx.x & 31;
  const int half = lane >> 4;
  const int mn   = lane & 15;
  const bf16_t* ar = A  + (size_t)(row0 + mn) * lda;
  const bf16_t* br = Bt + (size_t)(col0 + mn) * ldb;
  v8f acc = {0.f, 0.f, 0.f, 0.f, 0.f, 0.f, 0.f, 0.f};
  for (int k0 = 0; k0 < K; k0 += 32) {
    Frag a, b;
    a.h[0] = load_chunk(ar, k0 + 8 * half,       K);
    a.h[1] = load_chunk(ar, k0 + 16 + 8 * half,  K);
    b.h[0] = load_chunk(br, k0 + 8 * half,       K);
    b.h[1] = load_chunk(br, k0 + 16 + 8 * half,  K);
    acc = __builtin_amdgcn_wmma_f32_16x16x32_bf16(false, a.v, false, b.v,
                                                  (short)0, acc, false, false);
  }
  return acc;
}

// ---------------------------------------------------------------------------
// Stage 1: clustering
// ---------------------------------------------------------------------------

// Per-column mean / (std(ddof=1)+1e-6) for feats (384 cols) and coords (2 cols)
__global__ void hcla_colstats(const float* __restrict__ x, const float* __restrict__ coords,
                              float* __restrict__ stats) {
  __shared__ float sh0[256], sh1[256];
  int c = blockIdx.x, t = threadIdx.x;
  float s = 0.f, s2 = 0.f;
  if (c < CD) {
    for (int r = t; r < LT; r += 256) { float v = x[r * CD + c]; s += v; s2 += v * v; }
  } else {
    int cc = c - CD;
    for (int r = t; r < LT; r += 256) { float v = coords[r * 2 + cc]; s += v; s2 += v * v; }
  }
  sh0[t] = s; sh1[t] = s2; __syncthreads();
  for (int o = 128; o > 0; o >>= 1) {
    if (t < o) { sh0[t] += sh0[t + o]; sh1[t] += sh1[t + o]; }
    __syncthreads();
  }
  if (t == 0) {
    float n = (float)LT;
    float mu  = sh0[0] / n;
    float var = fmaxf((sh1[0] - mu * sh0[0]) / (n - 1.f), 0.f);
    stats[c * 2]     = mu;
    stats[c * 2 + 1] = sqrtf(var) + 1e-6f;
  }
}

// Standardize rows, row-normalize -> fzb (bf16); also x -> xb (bf16), cz (f32)
__global__ void hcla_rownorm(const float* __restrict__ x, const float* __restrict__ coords,
                             const float* __restrict__ stats,
                             bf16_t* __restrict__ fzb, bf16_t* __restrict__ xb,
                             float* __restrict__ cz) {
  __shared__ float tv[CD];
  __shared__ float red[128];
  int i = blockIdx.x, t = threadIdx.x;
  float acc = 0.f;
  for (int c = t; c < CD; c += 128) {
    float v = x[i * CD + c];
    xb[i * CD + c] = (bf16_t)v;
    float z = (v - stats[c * 2]) / stats[c * 2 + 1];
    tv[c] = z;
    acc += z * z;
  }
  red[t] = acc; __syncthreads();
  for (int o = 64; o > 0; o >>= 1) { if (t < o) red[t] += red[t + o]; __syncthreads(); }
  float inv = 1.f / fmaxf(sqrtf(red[0]), 1e-12f);
  for (int c = t; c < CD; c += 128) fzb[i * CD + c] = (bf16_t)(tv[c] * inv);
  if (t < 2) cz[i * 2 + t] = (coords[i * 2 + t] - stats[(CD + t) * 2]) / stats[(CD + t) * 2 + 1];
}

// Partial sum/sumsq of pairwise spatial distances (deterministic 2-stage)
__global__ void hcla_diststats(const float* __restrict__ cz, double* __restrict__ partial) {
  __shared__ double sh0[256], sh1[256];
  int t = threadIdx.x;
  double s = 0.0, s2 = 0.0;
  const long long total = (long long)LT * LT;
  for (long long idx = (long long)blockIdx.x * 256 + t; idx < total;
       idx += (long long)gridDim.x * 256) {
    int i = (int)(idx >> 12), j = (int)(idx & (LT - 1));
    float dx = cz[i * 2] - cz[j * 2], dy = cz[i * 2 + 1] - cz[j * 2 + 1];
    float d = sqrtf(dx * dx + dy * dy);
    s += d; s2 += (double)d * (double)d;
  }
  sh0[t] = s; sh1[t] = s2; __syncthreads();
  for (int o = 128; o > 0; o >>= 1) {
    if (t < o) { sh0[t] += sh0[t + o]; sh1[t] += sh1[t + o]; }
    __syncthreads();
  }
  if (t == 0) { partial[blockIdx.x * 2] = sh0[0]; partial[blockIdx.x * 2 + 1] = sh1[0]; }
}

// sigma (ddof=1) + softmax of weight_params
__global__ void hcla_finalize(const double* __restrict__ partial, const float* __restrict__ wp,
                              float* __restrict__ params) {
  double s = 0.0, s2 = 0.0;
  for (int b = 0; b < 1024; b++) { s += partial[b * 2]; s2 += partial[b * 2 + 1]; }
  double n = (double)LT * (double)LT;
  double mu = s / n;
  double var = (s2 - mu * s) / (n - 1.0);
  if (var < 0.0) var = 0.0;
  float sigma = (float)sqrt(var) + 1e-6f;
  float m = fmaxf(wp[0], wp[1]);
  float e0 = __expf(wp[0] - m), e1 = __expf(wp[1] - m);
  params[0] = e0 / (e0 + e1);
  params[1] = e1 / (e0 + e1);
  params[2] = 1.f / sigma;
}

// Fused: cos tile via WMMA + spatial kernel into 256KB LDS row-block,
// then wave-cooperative top-10 + row normalization.
__global__ void __launch_bounds__(256)
hcla_simtopk(const bf16_t* __restrict__ fzb, const float* __restrict__ cz,
             const float* __restrict__ params, float* __restrict__ knn) {
  extern __shared__ char smem[];
  float* sim = (float*)smem;                       // [16][4096]
  int t = threadIdx.x, wave = t >> 5, lane = t & 31, half = lane >> 4, mn = lane & 15;
  int row0 = blockIdx.x * 16;
  float w0 = params[0], w1 = params[1], isig = params[2];

  for (int tc = wave; tc < LT / 16; tc += 8) {
    int col0 = tc * 16;
    v8f acc = wmma_tile_nt(fzb, CD, fzb, CD, row0, col0, CD);
    int j = col0 + mn;
    float cjx = cz[j * 2], cjy = cz[j * 2 + 1];
#pragma unroll
    for (int v = 0; v < 8; v++) {
      int m = v + 8 * half;
      int i = row0 + m;
      float dx = cz[i * 2] - cjx, dy = cz[i * 2 + 1] - cjy;
      float d = sqrtf(dx * dx + dy * dy);
      sim[m * LT + j] = w0 * acc[v] + w1 * __expf(-d * isig);
    }
  }
  __syncthreads();

  for (int rr = 0; rr < 2; rr++) {
    int m = wave * 2 + rr;
    float list[KNNK];
#pragma unroll
    for (int s = 0; s < KNNK; s++) list[s] = -3.4e38f;
    for (int cix = lane; cix < LT; cix += 32) {
      float v = sim[m * LT + cix];
      if (v > list[KNNK - 1]) {
        int p = KNNK - 1;
        while (p > 0 && v > list[p - 1]) { list[p] = list[p - 1]; --p; }
        list[p] = v;
      }
    }
    int ptr = 0;
    float top[KNNK];
    for (int s = 0; s < KNNK; s++) {
      float cur = (ptr < KNNK) ? list[ptr] : -3.4e38f;
      float mx = cur;
#pragma unroll
      for (int o = 16; o > 0; o >>= 1) mx = fmaxf(mx, __shfl_xor(mx, o, 32));
      unsigned long long bal = __ballot(cur == mx);
      int firstLane = __ffsll(bal) - 1;
      if (lane == firstLane) ptr++;
      top[s] = mx;
    }
    if (lane == 0) {
      float sum = 0.f;
      for (int s = 0; s < KNNK; s++) sum += top[s];
      float inv = 1.f / (sum + 1e-6f);
      for (int s = 0; s < KNNK; s++) knn[(row0 + m) * KNNK + s] = top[s] * inv;
    }
  }
}

// Deterministic persistent-workgroup KMeans: knn features fully LDS-resident.
__global__ void hcla_kmeans(const float* __restrict__ knn, int* __restrict__ labels) {
  extern __shared__ char smem[];
  float* kn   = (float*)smem;              // LT*KNNK
  float* cent = kn + LT * KNNK;            // NCL*KNNK
  int*   labL = (int*)(cent + NCL * KNNK); // LT
  int t = threadIdx.x;
  for (int idx = t; idx < LT * KNNK; idx += 256) kn[idx] = knn[idx];
  __syncthreads();
  if (t < NCL * KNNK) {                    // linspace(0,4095,16) -> i*273
    int c = t / KNNK, d = t % KNNK;
    cent[t] = kn[(c * 273) * KNNK + d];
  }
  __syncthreads();
  for (int it = 0; it <= 50; ++it) {
    for (int p = t; p < LT; p += 256) {
      float f[KNNK];
#pragma unroll
      for (int d = 0; d < KNNK; d++) f[d] = kn[p * KNNK + d];
      float best = 3.4e38f; int lab = 0;
      for (int c = 0; c < NCL; c++) {
        float dd = 0.f;
#pragma unroll
        for (int d = 0; d < KNNK; d++) { float df = f[d] - cent[c * KNNK + d]; dd += df * df; }
        if (dd < best) { best = dd; lab = c; }
      }
      if (it == 50) labels[p] = lab; else labL[p] = lab;
    }
    __syncthreads();
    if (it < 50 && t < NCL * KNNK) {
      int c = t / KNNK, d = t % KNNK;
      float s = 0.f; int n = 0;
      for (int p = 0; p < LT; p++)
        if (labL[p] == c) { s += kn[p * KNNK + d]; n++; }
      if (n > 0) cent[t] = s / (float)n;
    }
    __syncthreads();
  }
}

// ---------------------------------------------------------------------------
// Stage 2: cluster-local masked attention (WMMA GEMMs)
// ---------------------------------------------------------------------------

__global__ void hcla_f2b(const float* __restrict__ src, bf16_t* __restrict__ dst, int n) {
  int i = blockIdx.x * blockDim.x + threadIdx.x;
  if (i < n) dst[i] = (bf16_t)src[i];
}

// qkv[4096,1152] = xb @ w_in^T + b_in
__global__ void __launch_bounds__(256)
hcla_qkv(const bf16_t* __restrict__ xb, const bf16_t* __restrict__ wb,
         const float* __restrict__ bias, float* __restrict__ qkv) {
  int wave = threadIdx.x >> 5, lane = threadIdx.x & 31, half = lane >> 4, n = lane & 15;
  int row0 = blockIdx.x * 16;
  int col0 = (blockIdx.y * 8 + wave) * 16;
  v8f acc = wmma_tile_nt(xb, CD, wb, CD, row0, col0, CD);
  int col = col0 + n;
  float bv = bias[col];
#pragma unroll
  for (int v = 0; v < 8; v++) qkv[(row0 + v + 8 * half) * C3 + col] = acc[v] + bv;
}

// Vt[h][d][j] = V[j][h*48+d]  (so PV becomes an NT GEMM with contiguous B rows)
__global__ void hcla_buildvt(const float* __restrict__ qkv, bf16_t* __restrict__ vtb) {
  int i = blockIdx.x * blockDim.x + threadIdx.x;
  if (i >= NH * HD * LT) return;
  int h = i / (HD * LT);
  int r = i % (HD * LT);
  int d = r / LT, j = r % LT;
  vtb[i] = (bf16_t)qkv[j * C3 + 2 * CD + h * HD + d];
}

// S = (Q K^T)/sqrt(48) with cluster mask (-inf outside own cluster)
__global__ void __launch_bounds__(256)
hcla_scores(const bf16_t* __restrict__ qkvb, const int* __restrict__ labels,
            float* __restrict__ S, int h) {
  int wave = threadIdx.x >> 5, lane = threadIdx.x & 31, half = lane >> 4, n = lane & 15;
  int row0 = blockIdx.x * 16;
  int col0 = (blockIdx.y * 8 + wave) * 16;
  const bf16_t* A = qkvb + h * HD;        // Q rows, stride C3
  const bf16_t* B = qkvb + CD + h * HD;   // K rows, stride C3
  v8f acc = wmma_tile_nt(A, C3, B, C3, row0, col0, HD);
  int j = col0 + n;
  int lj = labels[j];
#pragma unroll
  for (int v = 0; v < 8; v++) {
    int i = row0 + v + 8 * half;
    float s = acc[v] * SCALE;
    if (labels[i] != lj) s = -3.0e38f;
    S[i * LT + j] = s;
  }
}

__global__ void hcla_softmax(const float* __restrict__ S, bf16_t* __restrict__ P) {
  __shared__ float red[256];
  int i = blockIdx.x, t = threadIdx.x;
  const float* row = S + (size_t)i * LT;
  float mx = -3.4e38f;
  for (int c = t; c < LT; c += 256) mx = fmaxf(mx, row[c]);
  red[t] = mx; __syncthreads();
  for (int o = 128; o > 0; o >>= 1) { if (t < o) red[t] = fmaxf(red[t], red[t + o]); __syncthreads(); }
  mx = red[0]; __syncthreads();
  float s = 0.f;
  for (int c = t; c < LT; c += 256) s += __expf(row[c] - mx);
  red[t] = s; __syncthreads();
  for (int o = 128; o > 0; o >>= 1) { if (t < o) red[t] += red[t + o]; __syncthreads(); }
  float inv = 1.f / red[0];
  bf16_t* prow = P + (size_t)i * LT;
  for (int c = t; c < LT; c += 256) prow[c] = (bf16_t)(__expf(row[c] - mx) * inv);
}

// attnO[:, h*48 .. h*48+47] = P @ V_h   (A = P bf16, Bt = Vt[h], K = 4096)
__global__ void __launch_bounds__(96)
hcla_pv(const bf16_t* __restrict__ P, const bf16_t* __restrict__ vtb,
        float* __restrict__ attnO, int h) {
  int wave = threadIdx.x >> 5, lane = threadIdx.x & 31, half = lane >> 4, n = lane & 15;
  int row0 = blockIdx.x * 16;
  int col0 = wave * 16;                    // 3 col tiles cover HD=48
  const bf16_t* B = vtb + (size_t)h * HD * LT;
  v8f acc = wmma_tile_nt(P, LT, B, LT, row0, col0, LT);
#pragma unroll
  for (int v = 0; v < 8; v++) {
    int i = row0 + v + 8 * half;
    attnO[i * CD + h * HD + col0 + n] = acc[v];
  }
}

// out-proj + bias + residual x
__global__ void __launch_bounds__(256)
hcla_proj(const bf16_t* __restrict__ aob, const bf16_t* __restrict__ wob,
          const float* __restrict__ bias, const float* __restrict__ x,
          float* __restrict__ hpre) {
  int wave = threadIdx.x >> 5, lane = threadIdx.x & 31, half = lane >> 4, n = lane & 15;
  int row0 = blockIdx.x * 16;
  int col0 = (blockIdx.y * 8 + wave) * 16;
  v8f acc = wmma_tile_nt(aob, CD, wob, CD, row0, col0, CD);
  int col = col0 + n;
  float bv = bias[col];
#pragma unroll
  for (int v = 0; v < 8; v++) {
    int i = row0 + v + 8 * half;
    hpre[i * CD + col] = acc[v] + bv + x[i * CD + col];
  }
}

__global__ void hcla_layernorm(const float* __restrict__ in, const float* __restrict__ g,
                               const float* __restrict__ b, float* __restrict__ out) {
  __shared__ float r0[128], r1[128];
  int i = blockIdx.x, t = threadIdx.x;
  const float* row = in + i * CD;
  float s = 0.f, s2 = 0.f;
  for (int c = t; c < CD; c += 128) { float v = row[c]; s += v; s2 += v * v; }
  r0[t] = s; r1[t] = s2; __syncthreads();
  for (int o = 64; o > 0; o >>= 1) { if (t < o) { r0[t] += r0[t + o]; r1[t] += r1[t + o]; } __syncthreads(); }
  float mu  = r0[0] / (float)CD;
  float var = r1[0] / (float)CD - mu * mu;
  float inv = rsqrtf(var + LNEPS);
  for (int c = t; c < CD; c += 128) out[i * CD + c] = (row[c] - mu) * inv * g[c] + b[c];
}

// ---------------------------------------------------------------------------
// Stage 3: cluster reps + tiny global attention
// ---------------------------------------------------------------------------

__global__ void hcla_reps(const float* __restrict__ hmat, const int* __restrict__ labels,
                          float* __restrict__ reps) {
  int c = blockIdx.x;    // cluster
  int ch = threadIdx.x;  // channel (384 threads)
  float s = 0.f; int n = 0;
  for (int i = 0; i < LT; i++)
    if (labels[i] == c) { s += hmat[i * CD + ch]; n++; }
  reps[c * CD + ch] = s / fmaxf((float)n, 1.f);
}

// tiny NT GEMM: C[M,N] = A@W^T + bias (+optional residual R)
__global__ void hcla_small_nt(const float* __restrict__ A, const float* __restrict__ W,
                              const float* __restrict__ bias, const float* __restrict__ R,
                              float* __restrict__ Cm, int M, int N, int K) {
  int idx = blockIdx.x * blockDim.x + threadIdx.x;
  if (idx >= M * N) return;
  int i = idx / N, n = idx % N;
  float s = bias[n];
  for (int k = 0; k < K; k++) s += A[i * K + k] * W[n * K + k];
  if (R) s += R[idx];
  Cm[idx] = s;
}

__global__ void hcla_small_attn(const float* __restrict__ qkv2, float* __restrict__ out2) {
  __shared__ float sc[NCL][NCL];
  int h = blockIdx.x, t = threadIdx.x;
  if (t < NCL * NCL) {
    int qi = t / NCL, kj = t % NCL;
    float s = 0.f;
    for (int d = 0; d < HD; d++)
      s += qkv2[qi * C3 + h * HD + d] * qkv2[kj * C3 + CD + h * HD + d];
    sc[qi][kj] = s * SCALE;
  }
  __syncthreads();
  if (t < NCL) {
    float mx = -3.4e38f;
    for (int j = 0; j < NCL; j++) mx = fmaxf(mx, sc[t][j]);
    float sum = 0.f;
    for (int j = 0; j < NCL; j++) { float e = __expf(sc[t][j] - mx); sc[t][j] = e; sum += e; }
    float inv = 1.f / sum;
    for (int j = 0; j < NCL; j++) sc[t][j] *= inv;
  }
  __syncthreads();
  for (int idx = t; idx < NCL * HD; idx += 256) {
    int qi = idx / HD, d = idx % HD;
    float s = 0.f;
    for (int j = 0; j < NCL; j++) s += sc[qi][j] * qkv2[j * C3 + 2 * CD + h * HD + d];
    out2[qi * CD + h * HD + d] = s;
  }
}

__global__ void hcla_colmean(const float* __restrict__ reps2, float* __restrict__ rmean) {
  int c = threadIdx.x;
  if (c < CD) {
    float s = 0.f;
    for (int r = 0; r < NCL; r++) s += reps2[r * CD + c];
    rmean[c] = s / (float)NCL;
  }
}

__global__ void hcla_addmean(float* __restrict__ out, const float* __restrict__ rmean) {
  int idx = blockIdx.x * blockDim.x + threadIdx.x;
  if (idx < LT * CD) out[idx] += rmean[idx % CD];
}

// ---------------------------------------------------------------------------
// Host launcher
// ---------------------------------------------------------------------------
extern "C" void kernel_launch(void* const* d_in, const int* in_sizes, int n_in,
                              void* d_out, int out_size, void* d_ws, size_t ws_size,
                              hipStream_t stream) {
  const float* x      = (const float*)d_in[0];
  const float* coords = (const float*)d_in[1];
  const float* wp     = (const float*)d_in[2];
  const float* lw_in  = (const float*)d_in[3];
  const float* lb_in  = (const float*)d_in[4];
  const float* lw_out = (const float*)d_in[5];
  const float* lb_out = (const float*)d_in[6];
  const float* lg     = (const float*)d_in[7];
  const float* lb     = (const float*)d_in[8];
  const float* gw_in  = (const float*)d_in[9];
  const float* gb_in  = (const float*)d_in[10];
  const float* gw_out = (const float*)d_in[11];
  const float* gb_out = (const float*)d_in[12];
  const float* gg     = (const float*)d_in[13];
  const float* gbv    = (const float*)d_in[14];
  float* out = (float*)d_out;
  (void)in_sizes; (void)n_in; (void)out_size; (void)ws_size;

  char* w = (char*)d_ws;
  size_t off = 0;
  auto carve = [&](size_t bytes) -> char* {
    char* p = w + off;
    off += (bytes + 255) & ~(size_t)255;
    return p;
  };
  bf16_t* fzb    = (bf16_t*)carve((size_t)LT * CD * 2);
  bf16_t* xb     = (bf16_t*)carve((size_t)LT * CD * 2);
  float*  cz     = (float*) carve((size_t)LT * 2 * 4);
  float*  stats  = (float*) carve((size_t)(CD + 2) * 2 * 4);
  double* partial= (double*)carve((size_t)1024 * 2 * 8);
  float*  params = (float*) carve(4 * 4);
  float*  knn    = (float*) carve((size_t)LT * KNNK * 4);
  int*    labels = (int*)   carve((size_t)LT * 4);
  bf16_t* winb   = (bf16_t*)carve((size_t)C3 * CD * 2);
  bf16_t* woutb  = (bf16_t*)carve((size_t)CD * CD * 2);
  float*  qkv    = (float*) carve((size_t)LT * C3 * 4);
  bf16_t* qkvb   = (bf16_t*)carve((size_t)LT * C3 * 2);
  bf16_t* vtb    = (bf16_t*)carve((size_t)NH * HD * LT * 2);
  float*  S      = (float*) carve((size_t)LT * LT * 4);
  bf16_t* P      = (bf16_t*)carve((size_t)LT * LT * 2);
  float*  attnO  = (float*) carve((size_t)LT * CD * 4);
  bf16_t* aob    = (bf16_t*)carve((size_t)LT * CD * 2);
  float*  hpre   = (float*) carve((size_t)LT * CD * 4);
  float*  reps   = (float*) carve((size_t)NCL * CD * 4);
  float*  qkv2   = (float*) carve((size_t)NCL * C3 * 4);
  float*  out2   = (float*) carve((size_t)NCL * CD * 4);
  float*  hpre2  = (float*) carve((size_t)NCL * CD * 4);
  float*  reps2  = (float*) carve((size_t)NCL * CD * 4);
  float*  rmean  = (float*) carve((size_t)CD * 4);

  // -- clustering --
  hcla_colstats<<<CD + 2, 256, 0, stream>>>(x, coords, stats);
  hcla_rownorm<<<LT, 128, 0, stream>>>(x, coords, stats, fzb, xb, cz);
  hcla_diststats<<<1024, 256, 0, stream>>>(cz, partial);
  hcla_finalize<<<1, 1, 0, stream>>>(partial, wp, params);
  hcla_simtopk<<<LT / 16, 256, (size_t)16 * LT * 4, stream>>>(fzb, cz, params, knn);
  size_t kmLds = (size_t)LT * KNNK * 4 + (size_t)NCL * KNNK * 4 + (size_t)LT * 4;
  hcla_kmeans<<<1, 256, kmLds, stream>>>(knn, labels);

  // -- local masked attention --
  hcla_f2b<<<(C3 * CD + 255) / 256, 256, 0, stream>>>(lw_in, winb, C3 * CD);
  hcla_f2b<<<(CD * CD + 255) / 256, 256, 0, stream>>>(lw_out, woutb, CD * CD);
  hcla_qkv<<<dim3(LT / 16, C3 / 128), 256, 0, stream>>>(xb, winb, lb_in, qkv);
  hcla_f2b<<<(LT * C3 + 255) / 256, 256, 0, stream>>>(qkv, qkvb, LT * C3);
  hcla_buildvt<<<(NH * HD * LT + 255) / 256, 256, 0, stream>>>(qkv, vtb);
  for (int h = 0; h < NH; ++h) {
    hcla_scores<<<dim3(LT / 16, LT / 128), 256, 0, stream>>>(qkvb, labels, S, h);
    hcla_softmax<<<LT, 256, 0, stream>>>(S, P);
    hcla_pv<<<LT / 16, 96, 0, stream>>>(P, vtb, attnO, h);
  }
  hcla_f2b<<<(LT * CD + 255) / 256, 256, 0, stream>>>(attnO, aob, LT * CD);
  hcla_proj<<<dim3(LT / 16, CD / 128), 256, 0, stream>>>(aob, woutb, lb_out, x, hpre);
  hcla_layernorm<<<LT, 128, 0, stream>>>(hpre, lg, lb, out);

  // -- reps + global attention --
  hcla_reps<<<NCL, CD, 0, stream>>>(out, labels, reps);
  hcla_small_nt<<<(NCL * C3 + 255) / 256, 256, 0, stream>>>(reps, gw_in, gb_in, nullptr,
                                                            qkv2, NCL, C3, CD);
  hcla_small_attn<<<NH, 256, 0, stream>>>(qkv2, out2);
  hcla_small_nt<<<(NCL * CD + 255) / 256, 256, 0, stream>>>(out2, gw_out, gb_out, reps,
                                                            hpre2, NCL, CD, CD);
  hcla_layernorm<<<NCL, 128, 0, stream>>>(hpre2, gg, gbv, reps2);
  hcla_colmean<<<1, CD, 0, stream>>>(reps2, rmean);
  hcla_addmean<<<(LT * CD + 255) / 256, 256, 0, stream>>>(out, rmean);
}
